// DynamicShortConvolution_89627377533672
// MI455X (gfx1250) — compile-verified, matrix operand order checked by reading, not run on hardware
//
#include <hip/hip_runtime.h>
#include <hip/hip_bf16.h>

// Shapes (fixed by the reference)
#define DIM   2048   // hidden size D, also K of both GEMMs
#define DG    2048   // generator hidden
#define TLEN  4096
#define BATCH 2
#define ROWS  (BATCH * TLEN)   // 8192
#define KOUT  (DIM * 4)        // 8192 flat-kernel columns

typedef __attribute__((ext_vector_type(16))) __bf16 v16bf;
typedef __attribute__((ext_vector_type(8)))  __bf16 v8bf;
typedef __attribute__((ext_vector_type(4)))  __bf16 v4bf;
typedef __attribute__((ext_vector_type(8)))  float  v8f;
typedef __attribute__((ext_vector_type(4)))  float  v4f;

// Fast silu: x * rcp(1+exp(-x)) — v_exp_f32 + v_rcp_f32, no IEEE-div expansion.
__device__ __forceinline__ float silu_f(float x) {
    return x * __builtin_amdgcn_rcpf(1.0f + __expf(-x));
}

__device__ __forceinline__ v16bf cat16(v8bf lo, v8bf hi) {
    return __builtin_shufflevector(lo, hi, 0,1,2,3,4,5,6,7,8,9,10,11,12,13,14,15);
}

// Load one 16-bit WMMA A/B fragment (per-lane: 8 bf16 @ K=h*8, 8 bf16 @ K=16+h*8)
__device__ __forceinline__ v16bf load_frag(const __bf16* __restrict__ p) {
    v8bf lo = *(const v8bf*)(p);
    v8bf hi = *(const v8bf*)(p + 16);
    return cat16(lo, hi);
}

#define WMMA_BF16(a, b, c) \
    __builtin_amdgcn_wmma_f32_16x16x32_bf16(false, (a), false, (b), (short)0, (c), false, false)

// ---------------------------------------------------------------------------
// Kernel 0: fp32 -> bf16 conversion (vectorized x4)
// ---------------------------------------------------------------------------
__global__ __launch_bounds__(256) void cvt_f32_bf16(const float* __restrict__ in,
                                                    __bf16* __restrict__ out, int n4) {
    int i = blockIdx.x * blockDim.x + threadIdx.x;
    if (i >= n4) return;
    v4f v = *(const v4f*)(in + (size_t)i * 4);
    v4bf o;
    o[0] = (__bf16)v[0]; o[1] = (__bf16)v[1]; o[2] = (__bf16)v[2]; o[3] = (__bf16)v[3];
    *(v4bf*)(out + (size_t)i * 4) = o;
}

// ---------------------------------------------------------------------------
// Kernel 1: H = silu(X @ W1^T), bf16 in/out, f32 accumulate via WMMA.
// Block = 256 threads = 8 waves. Wave tile 32(M) x 32(N) = 2x2 WMMA tiles:
// 8 b128 loads per 4 WMMAs. Block tile: 32 rows(t) x 256 cols(g).
// grid = (ROWS/32, DG/256).
// ---------------------------------------------------------------------------
__global__ __launch_bounds__(256) void gemm1_silu_bf16(const __bf16* __restrict__ Xb,
                                                       const __bf16* __restrict__ W1b,
                                                       __bf16* __restrict__ Hb) {
    const int lane = threadIdx.x & 31;
    const int wv   = threadIdx.x >> 5;
    const int hf   = lane >> 4;
    const int r16  = lane & 15;

    const int m0 = blockIdx.x * 32;            // first t row of block
    const int g0 = blockIdx.y * 256 + wv * 32; // first g col of this wave

    const __bf16* a0ptr = Xb  + (size_t)(m0 + r16)      * DIM + hf * 8;
    const __bf16* a1ptr = Xb  + (size_t)(m0 + 16 + r16) * DIM + hf * 8;
    const __bf16* b0ptr = W1b + (size_t)(g0 + r16)      * DIM + hf * 8;
    const __bf16* b1ptr = W1b + (size_t)(g0 + 16 + r16) * DIM + hf * 8;

    v8f c00 = {}, c01 = {}, c10 = {}, c11 = {};
#pragma unroll 2
    for (int k = 0; k < DIM; k += 32) {
        v16bf a0 = load_frag(a0ptr + k);
        v16bf b0 = load_frag(b0ptr + k);
        c00 = WMMA_BF16(a0, b0, c00);
        v16bf b1 = load_frag(b1ptr + k);
        c01 = WMMA_BF16(a0, b1, c01);
        v16bf a1 = load_frag(a1ptr + k);
        c10 = WMMA_BF16(a1, b0, c10);
        c11 = WMMA_BF16(a1, b1, c11);
    }

    // C layout: lane holds rows M = v + 8*hf, col N = r16 within each tile.
    const int gcol = g0 + r16;
#pragma unroll
    for (int v = 0; v < 8; ++v) {
        const size_t r0 = (size_t)(m0 + 8 * hf + v) * DG;
        const size_t r1 = (size_t)(m0 + 16 + 8 * hf + v) * DG;
        Hb[r0 + gcol]      = (__bf16)silu_f(c00[v]);
        Hb[r0 + gcol + 16] = (__bf16)silu_f(c01[v]);
        Hb[r1 + gcol]      = (__bf16)silu_f(c10[v]);
        Hb[r1 + gcol + 16] = (__bf16)silu_f(c11[v]);
    }
}

// ---------------------------------------------------------------------------
// Kernel 2: fused  flat = H @ W2^T + b2  ->  causal depthwise dynamic conv
//           -> silu -> Y.  flat is never materialized in HBM.
// Block = 256 threads = 8 waves. Wave tile 32(M) x 64(N) = 2x4 WMMA tiles:
// 12 b128 loads per 8 WMMAs. Block tile of flat: 32 rows(t) x 512 cols(k)
// == 32 t x 128 d-channels x 4 taps, staged in 64 KB LDS.
// grid = (ROWS/32, KOUT/512).
// ---------------------------------------------------------------------------
__global__ __launch_bounds__(256) void gemm2_conv_silu(const __bf16* __restrict__ Hb,
                                                       const __bf16* __restrict__ W2b,
                                                       const float* __restrict__ b2,
                                                       const float* __restrict__ X,
                                                       float* __restrict__ Y) {
    __shared__ float sflat[32][512];   // exactly 64 KB

    const int lane = threadIdx.x & 31;
    const int wv   = threadIdx.x >> 5;
    const int hf   = lane >> 4;
    const int r16  = lane & 15;

    const int m0 = blockIdx.x * 32;              // first t row of block
    const int c0 = blockIdx.y * 512 + wv * 64;   // first flat column of this wave

    const __bf16* a0ptr = Hb + (size_t)(m0 + r16)      * DG + hf * 8;
    const __bf16* a1ptr = Hb + (size_t)(m0 + 16 + r16) * DG + hf * 8;
    const __bf16* bptr0 = W2b + (size_t)(c0 + r16)      * DG + hf * 8;
    const __bf16* bptr1 = W2b + (size_t)(c0 + 16 + r16) * DG + hf * 8;
    const __bf16* bptr2 = W2b + (size_t)(c0 + 32 + r16) * DG + hf * 8;
    const __bf16* bptr3 = W2b + (size_t)(c0 + 48 + r16) * DG + hf * 8;

    v8f c00 = {}, c01 = {}, c02 = {}, c03 = {};
    v8f c10 = {}, c11 = {}, c12 = {}, c13 = {};
#pragma unroll 2
    for (int k = 0; k < DG; k += 32) {
        v16bf a0 = load_frag(a0ptr + k);
        v16bf a1 = load_frag(a1ptr + k);
        v16bf b0 = load_frag(bptr0 + k);
        c00 = WMMA_BF16(a0, b0, c00);
        c10 = WMMA_BF16(a1, b0, c10);
        v16bf b1 = load_frag(bptr1 + k);
        c01 = WMMA_BF16(a0, b1, c01);
        c11 = WMMA_BF16(a1, b1, c11);
        v16bf b2f = load_frag(bptr2 + k);
        c02 = WMMA_BF16(a0, b2f, c02);
        c12 = WMMA_BF16(a1, b2f, c12);
        v16bf b3 = load_frag(bptr3 + k);
        c03 = WMMA_BF16(a0, b3, c03);
        c13 = WMMA_BF16(a1, b3, c13);
    }

    // Bias + stage flat tile into LDS.
    const int colloc = wv * 64 + r16;
    {
        const float bias0 = b2[c0 + r16];
        const float bias1 = b2[c0 + 16 + r16];
        const float bias2 = b2[c0 + 32 + r16];
        const float bias3 = b2[c0 + 48 + r16];
#pragma unroll
        for (int v = 0; v < 8; ++v) {
            float* row0 = &sflat[8 * hf + v][0];
            float* row1 = &sflat[16 + 8 * hf + v][0];
            row0[colloc]      = c00[v] + bias0;
            row0[colloc + 16] = c01[v] + bias1;
            row0[colloc + 32] = c02[v] + bias2;
            row0[colloc + 48] = c03[v] + bias3;
            row1[colloc]      = c10[v] + bias0;
            row1[colloc + 16] = c11[v] + bias1;
            row1[colloc + 32] = c12[v] + bias2;
            row1[colloc + 48] = c13[v] + bias3;
        }
    }
    __syncthreads();

    // Depthwise causal dynamic conv + silu.
    // 32 t x 128 d = 4096 outputs, 16 per thread. flat col k = d_local*4 + tap.
#pragma unroll
    for (int p = 0; p < 16; ++p) {
        const int idx  = threadIdx.x + p * 256;
        const int tt   = idx >> 7;           // 0..31  (t within tile)
        const int dd   = idx & 127;          // 0..127 (d within tile)
        const int trow = m0 + tt;
        const int bb   = trow >> 12;         // /4096 -> batch
        const int t    = trow & (TLEN - 1);  // t within batch (padding is per-batch)
        const int d    = blockIdx.y * 128 + dd;

        const float4 kv = *(const float4*)&sflat[tt][dd * 4];
        const float* xcol = X + ((size_t)bb * TLEN + t) * DIM + d;

        float y = 0.0f;
        if (t >= 3) y  = xcol[-3 * DIM] * kv.x;
        if (t >= 2) y += xcol[-2 * DIM] * kv.y;
        if (t >= 1) y += xcol[-1 * DIM] * kv.z;
        y += xcol[0] * kv.w;

        Y[((size_t)bb * TLEN + t) * DIM + d] = silu_f(y);
    }
}

// ---------------------------------------------------------------------------
extern "C" void kernel_launch(void* const* d_in, const int* in_sizes, int n_in,
                              void* d_out, int out_size, void* d_ws, size_t ws_size,
                              hipStream_t stream) {
    const float* x  = (const float*)d_in[0];   // [2,4096,2048]
    const float* w1 = (const float*)d_in[1];   // [2048,2048]
    const float* w2 = (const float*)d_in[2];   // [8192,2048]
    const float* b2 = (const float*)d_in[3];   // [8192]
    float* out      = (float*)d_out;           // [2,4096,2048]

    // Workspace layout (bf16): Xb | W1b | W2b | Hb  = 32+8+32+32 MiB
    char* ws = (char*)d_ws;
    __bf16* Xb  = (__bf16*)(ws);
    __bf16* W1b = (__bf16*)(ws + (size_t)ROWS * DIM * 2);
    __bf16* W2b = (__bf16*)(ws + (size_t)ROWS * DIM * 2 + (size_t)DG * DIM * 2);
    __bf16* Hb  = (__bf16*)(ws + (size_t)ROWS * DIM * 2 + (size_t)DG * DIM * 2
                               + (size_t)KOUT * DG * 2);

    // 0) fp32 -> bf16 conversions
    {
        int n4 = (ROWS * DIM) / 4;   // X
        cvt_f32_bf16<<<(n4 + 255) / 256, 256, 0, stream>>>(x, Xb, n4);
        n4 = (DG * DIM) / 4;         // W1
        cvt_f32_bf16<<<(n4 + 255) / 256, 256, 0, stream>>>(w1, W1b, n4);
        n4 = (KOUT * DG) / 4;        // W2
        cvt_f32_bf16<<<(n4 + 255) / 256, 256, 0, stream>>>(w2, W2b, n4);
    }

    // 1) H = silu(X @ W1^T)
    {
        dim3 grid(ROWS / 32, DG / 256);
        gemm1_silu_bf16<<<grid, 256, 0, stream>>>(Xb, W1b, Hb);
    }

    // 2) fused flat-GEMM + dynamic conv + silu
    {
        dim3 grid(ROWS / 32, KOUT / 512);
        gemm2_conv_silu<<<grid, 256, 0, stream>>>(Hb, W2b, b2, x, out);
    }
}
